// QuantizedLinear_76768245449069
// MI455X (gfx1250) — matrix operand order repaired and checked by reading, compiled, not run on hardware
//
#include <hip/hip_runtime.h>
#include <hip/hip_bf16.h>

typedef __attribute__((ext_vector_type(16))) _Float16 v16h;
typedef __attribute__((ext_vector_type(8)))  _Float16 v8h;
typedef __attribute__((ext_vector_type(8)))  float    v8f;
typedef __attribute__((ext_vector_type(8)))  int      v8i;

#define IN_F   4096
#define OUT_F  4096
#define BATCH  32
// per-wave K chunk = 512 (4 groups of 128), 8 waves cover K=4096

// ---- pre-pass: convert x (fp32) -> f16 in workspace (exact-ish, removes cvt from hot loop)
__global__ __launch_bounds__(256) void x_to_f16_kernel(const float* __restrict__ x,
                                                       _Float16* __restrict__ xh, int n) {
    int i = blockIdx.x * blockDim.x + threadIdx.x;
    if (i < n) xh[i] = (_Float16)x[i];
}

// ---- main kernel: one block per 16-col N tile; 8 waves split K; WMMA f16->f32
__global__ __launch_bounds__(256) void qlinear_wmma_kernel(
    const _Float16* __restrict__ xh,      // 32 x 4096 f16
    const int*      __restrict__ qw,      // 4096 x 4096 int32 in [-8,7]
    const float*    __restrict__ scales,  // 131072 group scales
    const float*    __restrict__ bias,    // 4096
    float*          __restrict__ out)     // 32 x 4096 fp32
{
    __shared__ float red[8 * 2 * 8 * 32];   // 8 waves x 2 M-tiles x 8 regs x 32 lanes = 16 KB

    const int tid      = threadIdx.x;
    const int wave     = tid >> 5;
    const int lane     = tid & 31;
    const int lane_mod = lane & 15;         // N (for B/C/D) or M-in-tile (for A)
    const int lane_hi  = lane >> 4;         // which K half of the fragment
    const int ntile    = blockIdx.x << 4;   // 16 output features per block
    const int n        = ntile + lane_mod;

    const int*      qrow  = qw + (size_t)n * IN_F;
    const _Float16* xrow0 = xh + (size_t)lane_mod * IN_F;          // M tile 0: rows 0..15
    const _Float16* xrow1 = xh + (size_t)(16 + lane_mod) * IN_F;   // M tile 1: rows 16..31

    const int kbase = wave << 9;            // wave * 512

    v8f acc0 = {};
    v8f acc1 = {};

#pragma unroll
    for (int g = 0; g < 4; ++g) {           // 4 groups of 128 per wave
        v8f p0 = {};
        v8f p1 = {};
        const int kg = kbase + (g << 7);

#pragma unroll
        for (int s = 0; s < 4; ++s) {       // 4 x K=32 WMMA per group
            const int k0 = kg + (s << 5);

            // B fragment: lane holds col n, K = k0 + 16*lane_hi + [0..15], contiguous int32
            const int kb = k0 + (lane_hi << 4);
            v8i q0 = *(const v8i*)(qrow + kb);
            v8i q1 = *(const v8i*)(qrow + kb + 8);
            v16h b;
#pragma unroll
            for (int i = 0; i < 8; ++i) {
                b[i]     = (_Float16)q0[i];   // int4 values are exact in f16
                b[i + 8] = (_Float16)q1[i];
            }

            // A fragments: lane holds row m, K = k0 + 16*lane_hi + [0..15], contiguous
            const int ka = k0 + (lane_hi << 4);
            v8h a0lo = *(const v8h*)(xrow0 + ka);
            v8h a0hi = *(const v8h*)(xrow0 + ka + 8);
            v8h a1lo = *(const v8h*)(xrow1 + ka);
            v8h a1hi = *(const v8h*)(xrow1 + ka + 8);
            v16h a0 = __builtin_shufflevector(a0lo, a0hi, 0,1,2,3,4,5,6,7,8,9,10,11,12,13,14,15);
            v16h a1 = __builtin_shufflevector(a1lo, a1hi, 0,1,2,3,4,5,6,7,8,9,10,11,12,13,14,15);

            p0 = __builtin_amdgcn_wmma_f32_16x16x32_f16(false, a0, false, b, (short)0, p0, false, false);
            p1 = __builtin_amdgcn_wmma_f32_16x16x32_f16(false, a1, false, b, (short)0, p1, false, false);
        }

        // group-wise dequant: scale is per (n, group); n = lane%16 is lane-uniform across
        // the fragment, so this is a per-lane scalar FMA into the fp32 accumulator.
        const float sc = scales[(size_t)n * 32 + (kbase >> 7) + g];
        acc0 += p0 * sc;
        acc1 += p1 * sc;
    }

    // ---- cross-wave K reduction in LDS
    {
        float* m0 = &red[(wave * 2 + 0) * 256];
        float* m1 = &red[(wave * 2 + 1) * 256];
#pragma unroll
        for (int r = 0; r < 8; ++r) {
            m0[r * 32 + lane] = acc0[r];
            m1[r * 32 + lane] = acc1[r];
        }
    }
    __syncthreads();

    // 256 threads: each sums one (reg,lane) slot over 8 waves, for both M tiles
    const int r = tid >> 5;
    const int l = tid & 31;
#pragma unroll
    for (int t = 0; t < 2; ++t) {
        float s = 0.0f;
#pragma unroll
        for (int w = 0; w < 8; ++w)
            s += red[((w * 2 + t) * 8 + r) * 32 + l];
        // C/D layout: lanes 0-15 -> M=r, N=l; lanes 16-31 -> M=r+8, N=l-16
        const int Ml = (l < 16) ? r : (r + 8);
        const int Nl = l & 15;
        const int m  = t * 16 + Ml;
        const int nn = ntile + Nl;
        out[(size_t)m * OUT_F + nn] = s + bias[nn];
    }
}

extern "C" void kernel_launch(void* const* d_in, const int* in_sizes, int n_in,
                              void* d_out, int out_size, void* d_ws, size_t ws_size,
                              hipStream_t stream) {
    const float* x      = (const float*)d_in[0];
    const int*   qw     = (const int*)d_in[1];
    const float* scales = (const float*)d_in[2];
    const float* bias   = (const float*)d_in[3];
    float*       out    = (float*)d_out;

    _Float16* xh = (_Float16*)d_ws;              // 32*4096 f16 = 256 KB scratch
    const int nx = BATCH * IN_F;

    x_to_f16_kernel<<<nx / 256, 256, 0, stream>>>(x, xh, nx);
    qlinear_wmma_kernel<<<OUT_F / 16, 256, 0, stream>>>(xh, qw, scales, bias, out);
}
